// BasicTransformerBlock_56238301773984
// MI455X (gfx1250) — compile-verified, matrix operand order checked by reading, not run on hardware
//
#include <hip/hip_runtime.h>
#include <hip/hip_bf16.h>
#include <math.h>

typedef __attribute__((ext_vector_type(16))) _Float16 v16h;
typedef __attribute__((ext_vector_type(8)))  _Float16 v8h;
typedef __attribute__((ext_vector_type(8)))  float    v8f;

#define LDS_STRIDE 40   // 32 halfs + 8 pad (80B rows, 16B-aligned fragment reads)
#define BM 128
#define BN 128
#define BK 32

// ---------------------------------------------------------------------------
// fp32 [K][N] -> f16 transposed [N][K]
// ---------------------------------------------------------------------------
__global__ void cvt_transpose_kernel(const float* __restrict__ in, _Float16* __restrict__ out,
                                     int K, int N) {
    size_t idx = (size_t)blockIdx.x * blockDim.x + threadIdx.x;
    size_t total = (size_t)K * N;
    if (idx >= total) return;
    size_t n = idx / (size_t)K;
    size_t k = idx - n * (size_t)K;
    out[idx] = (_Float16)in[k * (size_t)N + n];
}

// fp32 -> f16 elementwise
__global__ void cvt_plain_kernel(const float* __restrict__ in, _Float16* __restrict__ out, size_t n) {
    size_t idx = (size_t)blockIdx.x * blockDim.x + threadIdx.x;
    if (idx < n) out[idx] = (_Float16)in[idx];
}

// ---------------------------------------------------------------------------
// LayerNorm: one 256-thread block per row, f16 output
// ---------------------------------------------------------------------------
__global__ __launch_bounds__(256)
void layernorm_f16_kernel(const float* __restrict__ x, const float* __restrict__ g,
                          const float* __restrict__ bb, _Float16* __restrict__ out, int Dd) {
    int row = blockIdx.x;
    const float* xr = x + (size_t)row * Dd;
    float s = 0.f, s2 = 0.f;
    for (int c = threadIdx.x; c < Dd; c += blockDim.x) { float v = xr[c]; s += v; s2 += v * v; }
    __shared__ float sh[256], sh2[256];
    sh[threadIdx.x] = s; sh2[threadIdx.x] = s2;
    __syncthreads();
    for (int st = 128; st > 0; st >>= 1) {
        if ((int)threadIdx.x < st) { sh[threadIdx.x] += sh[threadIdx.x + st]; sh2[threadIdx.x] += sh2[threadIdx.x + st]; }
        __syncthreads();
    }
    float mean = sh[0] / Dd;
    float var  = sh2[0] / Dd - mean * mean;
    float inv  = rsqrtf(var + 1e-12f);
    for (int c = threadIdx.x; c < Dd; c += blockDim.x) {
        float v = (xr[c] - mean) * inv * g[c] + bb[c];
        out[(size_t)row * Dd + c] = (_Float16)v;
    }
}

// ---------------------------------------------------------------------------
// WMMA GEMM:  C[M,N] = A[M,K](f16) @ Bt[N,K](f16)^T + epilogue
// Block tile 128x128, 256 threads = 8 wave32 (4x2), wave tile 32x64:
//   2 A-frags x 4 B-frags -> 8 WMMAs per 32-K step, double-buffered LDS,
//   register-staged global prefetch, one barrier per K-step.
// Epilogue modes:
//   0: out16[m,n] = c                          (Q/K projections)
//   1: Vt scatter: out16[((b*H+h)*DH+dh)*T+t]  (V projection, per-head transpose)
//   2: outf[m,n]  = c + bias[n] + base[m,n]    (O-proj residual, FF2 final)
//   3: out16[m,n] = c + bias[n]                (FF h-half)
//   4: out16[m,n] = gelu(c+bias[n]) * yh[m,n]  (FF gate, in-place over yh)
// ---------------------------------------------------------------------------
__global__ __launch_bounds__(256)
void gemm_wmma_kernel(const _Float16* __restrict__ A, const _Float16* __restrict__ Bt,
                      int M, int N, int K, int mode,
                      const float* __restrict__ bias, const float* __restrict__ base,
                      float* __restrict__ outf, _Float16* __restrict__ out16,
                      const _Float16* __restrict__ yh, int Tv) {
    __shared__ _Float16 As[2][BM * LDS_STRIDE];
    __shared__ _Float16 Bs[2][BN * LDS_STRIDE];

    const int m0 = blockIdx.y * BM;
    const int n0 = blockIdx.x * BN;
    const int tid  = threadIdx.x;
    const int lane = tid & 31;
    const int wave = tid >> 5;
    const int lo = lane & 15, hi = lane >> 4;
    const int kc = hi * 8;
    const int msub = (wave & 3) * 32;   // 4 wave-rows  -> 128 M
    const int nsub = (wave >> 2) * 64;  // 2 wave-cols  -> 128 N

    // global staging coords: 512 16B-chunks per 128x32 tile, 2 per thread
    int ldrow[2], ldcol[2];
#pragma unroll
    for (int i = 0; i < 2; ++i) { int c = tid + i * 256; ldrow[i] = c >> 2; ldcol[i] = (c & 3) * 8; }

    v8f acc[2][4] = {};
    v8h ga[2], gb[2];

    // prologue: stage k0 = 0
#pragma unroll
    for (int i = 0; i < 2; ++i) {
        ga[i] = *(const v8h*)(A  + (size_t)(m0 + ldrow[i]) * K + ldcol[i]);
        gb[i] = *(const v8h*)(Bt + (size_t)(n0 + ldrow[i]) * K + ldcol[i]);
    }

    int buf = 0;
    for (int k0 = 0; k0 < K; k0 += BK) {
        // commit staged tile to LDS[buf]
#pragma unroll
        for (int i = 0; i < 2; ++i) {
            *(v8h*)&As[buf][ldrow[i] * LDS_STRIDE + ldcol[i]] = ga[i];
            *(v8h*)&Bs[buf][ldrow[i] * LDS_STRIDE + ldcol[i]] = gb[i];
        }
        __syncthreads();

        // prefetch next tile while computing
        int kn = k0 + BK;
        if (kn < K) {
#pragma unroll
            for (int i = 0; i < 2; ++i) {
                ga[i] = *(const v8h*)(A  + (size_t)(m0 + ldrow[i]) * K + kn + ldcol[i]);
                gb[i] = *(const v8h*)(Bt + (size_t)(n0 + ldrow[i]) * K + kn + ldcol[i]);
            }
        }

        v16h a[2], b[4];
#pragma unroll
        for (int im = 0; im < 2; ++im) {
            const _Float16* p = &As[buf][(msub + im * 16 + lo) * LDS_STRIDE + kc];
            *(v8h*)&a[im]       = *(const v8h*)p;
            *((v8h*)&a[im] + 1) = *(const v8h*)(p + 16);
        }
#pragma unroll
        for (int jn = 0; jn < 4; ++jn) {
            const _Float16* p = &Bs[buf][(nsub + jn * 16 + lo) * LDS_STRIDE + kc];
            *(v8h*)&b[jn]       = *(const v8h*)p;
            *((v8h*)&b[jn] + 1) = *(const v8h*)(p + 16);
        }
#pragma unroll
        for (int im = 0; im < 2; ++im)
#pragma unroll
            for (int jn = 0; jn < 4; ++jn)
                acc[im][jn] = __builtin_amdgcn_wmma_f32_16x16x32_f16(
                    false, a[im], false, b[jn], (short)0, acc[im][jn], false, false);

        buf ^= 1;
    }

    // epilogue
#pragma unroll
    for (int im = 0; im < 2; ++im) {
#pragma unroll
        for (int jn = 0; jn < 4; ++jn) {
            int n = n0 + nsub + jn * 16 + lo;
#pragma unroll
            for (int r = 0; r < 8; ++r) {
                int m = m0 + msub + im * 16 + hi * 8 + r;
                float c = acc[im][jn][r];
                size_t idx = (size_t)m * N + n;
                if (mode == 0) {
                    out16[idx] = (_Float16)c;
                } else if (mode == 1) {
                    int bbi = m / Tv;  int t  = m - bbi * Tv;
                    int hh  = n >> 6;  int dh = n & 63;
                    out16[(((size_t)bbi * 16 + hh) * 64 + dh) * Tv + t] = (_Float16)c;
                } else if (mode == 2) {
                    outf[idx] = c + bias[n] + base[idx];
                } else if (mode == 3) {
                    out16[idx] = (_Float16)(c + bias[n]);
                } else { // mode 4: exact-gelu gate * h
                    float yv = c + bias[n];
                    float gg = 0.5f * yv * (1.0f + erff(yv * 0.70710678118654752f));
                    out16[idx] = (_Float16)(gg * (float)yh[idx]);
                }
            }
        }
    }
}

// ---------------------------------------------------------------------------
// Flash attention: one wave32 per (b, h, 16-query tile). DH=64, D=1024.
// Q,K f16 [B*S][1024]; Vt f16 [B][H][64][T]; O f16 [B*S][1024].
// 8 WMMAs per 32-key step (4 score + 4 P.V), online softmax via shfl_xor.
// ---------------------------------------------------------------------------
__global__ __launch_bounds__(32)
void flash_attn_kernel(const _Float16* __restrict__ Q, const _Float16* __restrict__ Kb,
                       const _Float16* __restrict__ Vt, const int* __restrict__ mask,
                       _Float16* __restrict__ O, int S, int T, float scale) {
    const int lane = threadIdx.x;
    const int lo = lane & 15, hi = lane >> 4, kc = hi * 8;
    const int b = blockIdx.z, h = blockIdx.y, q0 = blockIdx.x * 16;
    const int Dd = 1024;
    __shared__ _Float16 Ps[16 * LDS_STRIDE];

    // Q as two 16x32 A-fragments (dh 0..31, 32..63)
    const _Float16* qrow = Q + (size_t)(b * S + q0 + lo) * Dd + h * 64;
    v16h qa[2];
#pragma unroll
    for (int f = 0; f < 2; ++f) {
        *(v8h*)&qa[f]       = *(const v8h*)(qrow + f * 32 + kc);
        *((v8h*)&qa[f] + 1) = *(const v8h*)(qrow + f * 32 + kc + 16);
    }

    v8f acc[4] = {};
    float mrun[8], lrun[8];
#pragma unroll
    for (int r = 0; r < 8; ++r) { mrun[r] = -1e30f; lrun[r] = 0.f; }

    for (int kt = 0; kt < T; kt += 32) {
        // ---- scores: Q (16x64) x K^T -> 16x32 (two 16-key subtiles) ----
        v8f sc[2] = {};
#pragma unroll
        for (int s2 = 0; s2 < 2; ++s2) {
            const _Float16* krow = Kb + (size_t)(b * T + kt + s2 * 16 + lo) * Dd + h * 64;
#pragma unroll
            for (int f = 0; f < 2; ++f) {
                v16h kf;
                *(v8h*)&kf       = *(const v8h*)(krow + f * 32 + kc);
                *((v8h*)&kf + 1) = *(const v8h*)(krow + f * 32 + kc + 16);
                sc[s2] = __builtin_amdgcn_wmma_f32_16x16x32_f16(
                    false, qa[f], false, kf, (short)0, sc[s2], false, false);
            }
        }
        float madd0 = 0.f, madd1 = 0.f;
        if (mask) {
            madd0 = mask[b * T + kt + lo]      ? 0.f : -1e30f;
            madd1 = mask[b * T + kt + 16 + lo] ? 0.f : -1e30f;
        }
        float p0[8], p1[8];
#pragma unroll
        for (int r = 0; r < 8; ++r) { p0[r] = sc[0][r] * scale + madd0; p1[r] = sc[1][r] * scale + madd1; }

        // ---- online softmax (row reductions across 16 lanes of the half) ----
#pragma unroll
        for (int r = 0; r < 8; ++r) {
            float rmax = fmaxf(p0[r], p1[r]);
            for (int x = 8; x >= 1; x >>= 1) rmax = fmaxf(rmax, __shfl_xor(rmax, x, 32));
            float mnew = fmaxf(mrun[r], rmax);
            float corr = __expf(mrun[r] - mnew);
            p0[r] = __expf(p0[r] - mnew);
            p1[r] = __expf(p1[r] - mnew);
            float rs = p0[r] + p1[r];
            for (int x = 8; x >= 1; x >>= 1) rs += __shfl_xor(rs, x, 32);
            lrun[r] = lrun[r] * corr + rs;
            mrun[r] = mnew;
#pragma unroll
            for (int j = 0; j < 4; ++j) acc[j][r] *= corr;
        }

        // ---- relayout P (C-layout -> A-fragment) through LDS ----
        __syncthreads();
#pragma unroll
        for (int r = 0; r < 8; ++r) {
            Ps[(hi * 8 + r) * LDS_STRIDE + lo]      = (_Float16)p0[r];
            Ps[(hi * 8 + r) * LDS_STRIDE + 16 + lo] = (_Float16)p1[r];
        }
        __syncthreads();
        v16h pa;
        *(v8h*)&pa       = *(const v8h*)&Ps[lo * LDS_STRIDE + kc];
        *((v8h*)&pa + 1) = *(const v8h*)&Ps[lo * LDS_STRIDE + kc + 16];

        // ---- P (16x32) x V (32x64) via Vt rows ----
#pragma unroll
        for (int j = 0; j < 4; ++j) {
            const _Float16* vrow = Vt + ((size_t)(b * 16 + h) * 64 + j * 16 + lo) * T + kt;
            v16h vb;
            *(v8h*)&vb       = *(const v8h*)(vrow + kc);
            *((v8h*)&vb + 1) = *(const v8h*)(vrow + kc + 16);
            acc[j] = __builtin_amdgcn_wmma_f32_16x16x32_f16(
                false, pa, false, vb, (short)0, acc[j], false, false);
        }
    }

#pragma unroll
    for (int r = 0; r < 8; ++r) {
        float inv = 1.0f / lrun[r];
        size_t row = (size_t)(b * S + q0 + hi * 8 + r);
#pragma unroll
        for (int j = 0; j < 4; ++j)
            O[row * Dd + h * 64 + j * 16 + lo] = (_Float16)(acc[j][r] * inv);
    }
}

// ---------------------------------------------------------------------------
extern "C" void kernel_launch(void* const* d_in, const int* in_sizes, int n_in,
                              void* d_out, int out_size, void* d_ws, size_t ws_size,
                              hipStream_t stream) {
    (void)in_sizes; (void)n_in; (void)out_size; (void)ws_size;
    const int B = 4, S = 1024, T = 1024, D = 1024, Hh = 16, FF = 4096;

    const float* hidden = (const float*)d_in[0];
    const float* context= (const float*)d_in[1];
    const int*   kpm    = (const int*)  d_in[2];
    const float* g1 = (const float*)d_in[3],  *b1 = (const float*)d_in[4];
    const float* g2 = (const float*)d_in[5],  *b2 = (const float*)d_in[6];
    const float* g3 = (const float*)d_in[7],  *b3 = (const float*)d_in[8];
    const float* Wq1 = (const float*)d_in[9],  *Wk1 = (const float*)d_in[10];
    const float* Wv1 = (const float*)d_in[11], *Wo1 = (const float*)d_in[12];
    const float* bo1 = (const float*)d_in[13];
    const float* Wq2 = (const float*)d_in[14], *Wk2 = (const float*)d_in[15];
    const float* Wv2 = (const float*)d_in[16], *Wo2 = (const float*)d_in[17];
    const float* bo2 = (const float*)d_in[18];
    const float* Wff1 = (const float*)d_in[19], *bff1 = (const float*)d_in[20];
    const float* Wff2 = (const float*)d_in[21], *bff2 = (const float*)d_in[22];
    float* out = (float*)d_out;

    // ---- workspace layout ----
    char* wsb = (char*)d_ws;
    size_t off = 0;
    auto a16 = [&](size_t n) -> _Float16* { _Float16* p = (_Float16*)(wsb + off); off += ((n * 2 + 255) / 256) * 256; return p; };
    auto a32 = [&](size_t n) -> float*    { float*    p = (float*)(wsb + off);    off += ((n * 4 + 255) / 256) * 256; return p; };

    const size_t DD = (size_t)D * D;
    const size_t AD = (size_t)B * S * D;
    _Float16 *wq1t = a16(DD), *wk1t = a16(DD), *wv1t = a16(DD), *wo1t = a16(DD);
    _Float16 *wq2t = a16(DD), *wk2t = a16(DD), *wv2t = a16(DD), *wo2t = a16(DD);
    _Float16 *wff1t = a16((size_t)D * 2 * FF), *wff2t = a16((size_t)FF * D);
    _Float16 *nbuf = a16(AD), *ctx16 = a16(AD);
    _Float16 *qb = a16(AD), *kb = a16(AD), *vtb = a16(AD), *aob = a16(AD);
    _Float16 *yh = a16((size_t)B * S * FF);
    float    *h1 = a32(AD);

    auto cvtT = [&](const float* in, _Float16* o, int K, int N) {
        size_t n = (size_t)K * N;
        cvt_transpose_kernel<<<(unsigned)((n + 255) / 256), 256, 0, stream>>>(in, o, K, N);
    };
    auto gemm = [&](const _Float16* Aa, const _Float16* Bt, int M, int N, int K, int mode,
                    const float* bias, const float* base, float* of, _Float16* o16,
                    const _Float16* yhp) {
        dim3 g(N / BN, M / BM);
        gemm_wmma_kernel<<<g, 256, 0, stream>>>(Aa, Bt, M, N, K, mode, bias, base, of, o16, yhp, T);
    };

    // ---- weight & activation converts ----
    cvtT(Wq1, wq1t, D, D);  cvtT(Wk1, wk1t, D, D);  cvtT(Wv1, wv1t, D, D);  cvtT(Wo1, wo1t, D, D);
    cvtT(Wq2, wq2t, D, D);  cvtT(Wk2, wk2t, D, D);  cvtT(Wv2, wv2t, D, D);  cvtT(Wo2, wo2t, D, D);
    cvtT(Wff1, wff1t, D, 2 * FF);
    cvtT(Wff2, wff2t, FF, D);
    cvt_plain_kernel<<<(unsigned)((AD + 255) / 256), 256, 0, stream>>>(context, ctx16, AD);

    const int M4 = B * S;

    // ---- block 1: self-attention ----
    layernorm_f16_kernel<<<M4, 256, 0, stream>>>(hidden, g1, b1, nbuf, D);
    gemm(nbuf, wq1t, M4, D, D, 0, nullptr, nullptr, nullptr, qb,  nullptr);
    gemm(nbuf, wk1t, M4, D, D, 0, nullptr, nullptr, nullptr, kb,  nullptr);
    gemm(nbuf, wv1t, M4, D, D, 1, nullptr, nullptr, nullptr, vtb, nullptr);
    flash_attn_kernel<<<dim3(S / 16, Hh, B), 32, 0, stream>>>(qb, kb, vtb, nullptr, aob, S, T, 0.125f);
    gemm(aob, wo1t, M4, D, D, 2, bo1, hidden, h1, nullptr, nullptr);   // h1 = hidden + attn1

    // ---- block 2: cross-attention ----
    layernorm_f16_kernel<<<M4, 256, 0, stream>>>(h1, g2, b2, nbuf, D);
    gemm(nbuf,  wq2t, M4, D, D, 0, nullptr, nullptr, nullptr, qb,  nullptr);
    gemm(ctx16, wk2t, M4, D, D, 0, nullptr, nullptr, nullptr, kb,  nullptr);
    gemm(ctx16, wv2t, M4, D, D, 1, nullptr, nullptr, nullptr, vtb, nullptr);
    flash_attn_kernel<<<dim3(S / 16, Hh, B), 32, 0, stream>>>(qb, kb, vtb, kpm, aob, S, T, 0.125f);
    gemm(aob, wo2t, M4, D, D, 2, bo2, h1, out, nullptr, nullptr);      // out = h1 + attn2 (= h2)

    // ---- block 3: GEGLU FF ----
    layernorm_f16_kernel<<<M4, 256, 0, stream>>>(out, g3, b3, nbuf, D);
    gemm(nbuf, wff1t,                  M4, FF, D, 3, bff1,      nullptr, nullptr, yh, nullptr); // Yh
    gemm(nbuf, wff1t + (size_t)FF * D, M4, FF, D, 4, bff1 + FF, nullptr, nullptr, yh, yh);      // gelu(gate)*Yh
    gemm(yh,   wff2t, M4, D, FF, 2, bff2, out, out, nullptr, nullptr);  // out = h2 + ff
}
